// KnowledgeGuidedGraphTransformer_154618822907
// MI455X (gfx1250) — matrix operand order repaired and checked by reading, not compiled, not run on hardware
//
#include <hip/hip_runtime.h>
#include <hip/hip_bf16.h>
#include <math.h>

typedef _Float16 h16_t;
typedef __attribute__((ext_vector_type(8)))  _Float16 v8h;
typedef __attribute__((ext_vector_type(16))) _Float16 v16h;
typedef __attribute__((ext_vector_type(8)))  float    v8f;

#define DEV_INLINE __device__ __forceinline__

DEV_INLINE float gelu_exact(float x) {
    return 0.5f * x * (1.0f + erff(x * 0.70710678118654752440f));
}

DEV_INLINE void atomicMaxF32(float* addr, float val) {
    int* ai = (int*)addr;
    int old = *ai;
    while (__int_as_float(old) < val) {
        int assumed = old;
        old = atomicCAS(ai, assumed, __float_as_int(val));
        if (old == assumed) break;
    }
}

// ---------------------------------------------------------------------------
// Generic WMMA f16 GEMM:  C[M,OUT] = act(A[M,K] @ W[OUT,K]^T + bias)
// One wave computes a 16x64 output strip (4 WMMA 16x16x32 tiles).
// M % 16 == 0, K % 32 == 0, OUT % 64 == 0 (true for all uses here).
// ---------------------------------------------------------------------------
__global__ void gemm_wmma_f16(const h16_t* __restrict__ A,
                              const h16_t* __restrict__ W,
                              const float* __restrict__ bias,
                              float* __restrict__ C32,
                              h16_t* __restrict__ C16,
                              int M, int K, int OUT, int act)
{
    const int wave  = blockIdx.x * (blockDim.x >> 5) + (threadIdx.x >> 5);
    const int cgs   = OUT >> 6;                 // column groups of 64
    const int mtile = wave / cgs;
    const int cg    = wave - mtile * cgs;
    if (mtile * 16 >= M) return;

    const int lane = threadIdx.x & 31;
    const int hi   = lane >> 4;                 // half-wave select
    const int l16  = lane & 15;

    const h16_t* arow = A + (size_t)(mtile * 16 + l16) * K;
    const h16_t* b0p  = W + (size_t)(cg * 64 +  0 + l16) * K + hi * 16;
    const h16_t* b1p  = W + (size_t)(cg * 64 + 16 + l16) * K + hi * 16;
    const h16_t* b2p  = W + (size_t)(cg * 64 + 32 + l16) * K + hi * 16;
    const h16_t* b3p  = W + (size_t)(cg * 64 + 48 + l16) * K + hi * 16;

    v8f acc0 = {}, acc1 = {}, acc2 = {}, acc3 = {};

    for (int k0 = 0; k0 < K; k0 += 32) {
        // A fragment: 16 contiguous K halves, K = k0+16*hi .. +15
        v16h a  = *(const v16h*)(arow + k0 + hi * 16);
        // B fragments: lane holds col n = l16, 16 contiguous K halves
        v16h b0 = *(const v16h*)(b0p + k0);
        v16h b1 = *(const v16h*)(b1p + k0);
        v16h b2 = *(const v16h*)(b2p + k0);
        v16h b3 = *(const v16h*)(b3p + k0);
        acc0 = __builtin_amdgcn_wmma_f32_16x16x32_f16(false, a, false, b0, (short)0, acc0, false, false);
        acc1 = __builtin_amdgcn_wmma_f32_16x16x32_f16(false, a, false, b1, (short)0, acc1, false, false);
        acc2 = __builtin_amdgcn_wmma_f32_16x16x32_f16(false, a, false, b2, (short)0, acc2, false, false);
        acc3 = __builtin_amdgcn_wmma_f32_16x16x32_f16(false, a, false, b3, (short)0, acc3, false, false);
    }

    const int mbase = mtile * 16 + hi * 8;
    #define STORE_TILE(ACC, T)                                              \
    {                                                                       \
        const int n_out = cg * 64 + (T) * 16 + l16;                         \
        const float bv  = bias ? bias[n_out] : 0.0f;                        \
        _Pragma("unroll")                                                   \
        for (int r = 0; r < 8; ++r) {                                       \
            float val = ACC[r] + bv;                                        \
            if (act == 1) val = gelu_exact(val);                            \
            const size_t idx = (size_t)(mbase + r) * OUT + n_out;           \
            if (C32) C32[idx] = val;                                        \
            if (C16) C16[idx] = (h16_t)val;                                 \
        }                                                                   \
    }
    STORE_TILE(acc0, 0) STORE_TILE(acc1, 1) STORE_TILE(acc2, 2) STORE_TILE(acc3, 3)
    #undef STORE_TILE
}

// ---------------------------------------------------------------------------
// Elementwise / utility kernels
// ---------------------------------------------------------------------------
__global__ void fill_f32(float* __restrict__ p, float v, int n) {
    int i = blockIdx.x * blockDim.x + threadIdx.x;
    if (i < n) p[i] = v;
}

__global__ void convert_f16(const float* __restrict__ s, h16_t* __restrict__ d, int n) {
    int i = blockIdx.x * blockDim.x + threadIdx.x;
    if (i < n) d[i] = (h16_t)s[i];
}

__global__ void input_proj(const float* __restrict__ x, const float* __restrict__ w,
                           const float* __restrict__ b,
                           float* __restrict__ h, h16_t* __restrict__ hh, int total) {
    int i = blockIdx.x * blockDim.x + threadIdx.x;
    if (i >= total) return;
    int n = i >> 8, c = i & 255;
    float v = x[n] * w[c] + b[c];
    h[i] = v;
    hh[i] = (h16_t)v;
}

// ---------------------------------------------------------------------------
// Edge-feature gating MLP: gate[e,h] = sigmoid(relu(ea*w1+b1) @ w2[h] + b2[h])
// ---------------------------------------------------------------------------
__global__ void edge_gate(const float* __restrict__ ea,
                          const float* __restrict__ w1, const float* __restrict__ b1,
                          const float* __restrict__ w2, const float* __restrict__ b2,
                          float* __restrict__ gate, int E) {
    int e = blockIdx.x * blockDim.x + threadIdx.x;
    if (e >= E) return;
    float a = ea[e];
    float acc[8];
    #pragma unroll
    for (int h = 0; h < 8; ++h) acc[h] = b2[h];
    for (int j = 0; j < 64; ++j) {
        float eh = fmaxf(a * w1[j] + b1[j], 0.0f);
        #pragma unroll
        for (int h = 0; h < 8; ++h) acc[h] += eh * w2[h * 64 + j];
    }
    #pragma unroll
    for (int h = 0; h < 8; ++h)
        gate[(size_t)e * 8 + h] = 1.0f / (1.0f + expf(-acc[h]));
}

// One wave per edge: scores[e,h] = (q[src]·k[dst])_h / sqrt(32) * gate; segment max
__global__ void edge_scores(const float* __restrict__ q, const float* __restrict__ k,
                            const int* __restrict__ src, const int* __restrict__ dst,
                            const float* __restrict__ gate,
                            float* __restrict__ scores, float* __restrict__ smax, int E) {
    int e = blockIdx.x * (blockDim.x >> 5) + (threadIdx.x >> 5);
    if (e >= E) return;
    int lane = threadIdx.x & 31;
    int s = src[e], d = dst[e];
    const float* qp = q + (size_t)s * 256 + lane * 8;
    const float* kp = k + (size_t)d * 256 + lane * 8;
    float4 qa = *(const float4*)qp, qb = *(const float4*)(qp + 4);
    float4 ka = *(const float4*)kp, kb = *(const float4*)(kp + 4);
    float p = qa.x * ka.x + qa.y * ka.y + qa.z * ka.z + qa.w * ka.w
            + qb.x * kb.x + qb.y * kb.y + qb.z * kb.z + qb.w * kb.w;
    p += __shfl_xor(p, 1);
    p += __shfl_xor(p, 2);
    if ((lane & 3) == 0) {
        int h = lane >> 2;
        float sc = (p * 0.17677669529663687f) * gate[(size_t)e * 8 + h];
        scores[(size_t)e * 8 + h] = sc;
        atomicMaxF32(&smax[(size_t)d * 8 + h], sc);
    }
}

// ex = exp(score - smax[dst]); den += ex  (overwrites scores with ex)
__global__ void edge_exp(float* __restrict__ scores, const int* __restrict__ dst,
                         const float* __restrict__ smax, float* __restrict__ den, int total) {
    int i = blockIdx.x * blockDim.x + threadIdx.x;
    if (i >= total) return;
    int e = i >> 3, h = i & 7;
    int d = dst[e];
    float ex = expf(scores[i] - smax[(size_t)d * 8 + h]);
    scores[i] = ex;
    atomicAdd(&den[(size_t)d * 8 + h], ex);
}

// One wave per edge: agg[dst] += (ex/den[dst]) * v[src]
__global__ void edge_aggregate(const float* __restrict__ v, const float* __restrict__ ex,
                               const float* __restrict__ den,
                               const int* __restrict__ src, const int* __restrict__ dst,
                               float* __restrict__ agg, int E) {
    int e = blockIdx.x * (blockDim.x >> 5) + (threadIdx.x >> 5);
    if (e >= E) return;
    int lane = threadIdx.x & 31;
    int s = src[e], d = dst[e];
    int h = lane >> 2;
    float attn = ex[(size_t)e * 8 + h] / den[(size_t)d * 8 + h];
    const float* vp = v + (size_t)s * 256 + lane * 8;
    float* ap = agg + (size_t)d * 256 + lane * 8;
    float4 va = *(const float4*)vp, vb = *(const float4*)(vp + 4);
    atomicAdd(ap + 0, attn * va.x); atomicAdd(ap + 1, attn * va.y);
    atomicAdd(ap + 2, attn * va.z); atomicAdd(ap + 3, attn * va.w);
    atomicAdd(ap + 4, attn * vb.x); atomicAdd(ap + 5, attn * vb.y);
    atomicAdd(ap + 6, attn * vb.z); atomicAdd(ap + 7, attn * vb.w);
}

// ---------------------------------------------------------------------------
// Fused residual + LayerNorm, one wave per row (cols % 32 == 0, cols <= 256)
// ---------------------------------------------------------------------------
__global__ void ln_rows(const float* __restrict__ a, const float* __restrict__ r,
                        const float* __restrict__ w, const float* __restrict__ b,
                        float* __restrict__ y, h16_t* __restrict__ y16,
                        int rows, int cols) {
    int row = blockIdx.x * (blockDim.x >> 5) + (threadIdx.x >> 5);
    if (row >= rows) return;
    int lane = threadIdx.x & 31;
    int cpl  = cols >> 5;
    float vals[8];
    const float* ap = a + (size_t)row * cols;
    const float* rp = r ? r + (size_t)row * cols : nullptr;
    float sum = 0.0f;
    for (int j = 0; j < cpl; ++j) {
        int c = lane * cpl + j;
        float x = ap[c] + (rp ? rp[c] : 0.0f);
        vals[j] = x;
        sum += x;
    }
    for (int off = 16; off; off >>= 1) sum += __shfl_xor(sum, off);
    float mu = sum / (float)cols;
    float vs = 0.0f;
    for (int j = 0; j < cpl; ++j) { float d0 = vals[j] - mu; vs += d0 * d0; }
    for (int off = 16; off; off >>= 1) vs += __shfl_xor(vs, off);
    float rstd = rsqrtf(vs / (float)cols + 1e-5f);
    for (int j = 0; j < cpl; ++j) {
        int c = lane * cpl + j;
        float o = (vals[j] - mu) * rstd * w[c] + b[c];
        y[(size_t)row * cols + c] = o;
        if (y16) y16[(size_t)row * cols + c] = (h16_t)o;
    }
}

// ---------------------------------------------------------------------------
// Pooling and small residual blocks
// ---------------------------------------------------------------------------
__global__ void pool_count(const int* __restrict__ batch, float* __restrict__ counts, int n) {
    int i = blockIdx.x * blockDim.x + threadIdx.x;
    if (i < n) atomicAdd(&counts[batch[i]], 1.0f);
}

__global__ void pool_accum(const float* __restrict__ h, const int* __restrict__ batch,
                           float* __restrict__ acc, int total) {
    int i = blockIdx.x * blockDim.x + threadIdx.x;
    if (i >= total) return;
    int n = i >> 8, c = i & 255;
    atomicAdd(&acc[(size_t)batch[n] * 256 + c], h[i]);
}

__global__ void pool_final(const float* __restrict__ acc, const float* __restrict__ counts,
                           float* __restrict__ out_emb, float* __restrict__ embbuf, int total) {
    int i = blockIdx.x * blockDim.x + threadIdx.x;
    if (i >= total) return;
    int g = i >> 8;
    float v = acc[i] / counts[g];
    out_emb[i] = v;
    embbuf[i]  = v;
}

__global__ void small_linear(const float* __restrict__ A, const float* __restrict__ W,
                             const float* __restrict__ B, float* __restrict__ C,
                             int M, int K, int O, int act) {
    int i = blockIdx.x * blockDim.x + threadIdx.x;
    if (i >= M * O) return;
    int m = i / O, o = i - m * O;
    const float* ap = A + (size_t)m * K;
    const float* wp = W + (size_t)o * K;
    float acc = B ? B[o] : 0.0f;
    for (int j = 0; j < K; ++j) acc += ap[j] * wp[j];
    if (act) acc = gelu_exact(acc);
    C[i] = acc;
}

__global__ void heads_kernel(const float* __restrict__ f3,
                             const float* __restrict__ mw, const float* __restrict__ mb,
                             const float* __restrict__ vw, const float* __restrict__ vb,
                             float* __restrict__ out_mean, float* __restrict__ out_var) {
    int g = threadIdx.x;
    if (g >= 32) return;
    float m = mb[0], vv = vb[0];
    for (int j = 0; j < 64; ++j) {
        float t = f3[g * 64 + j];
        m  += t * mw[j];
        vv += t * vw[j];
    }
    out_mean[g] = m;
    out_var[g]  = expf(vv) + 1e-6f;
}

// ---------------------------------------------------------------------------
// Host launcher
// ---------------------------------------------------------------------------
extern "C" void kernel_launch(void* const* d_in, const int* in_sizes, int n_in,
                              void* d_out, int out_size, void* d_ws, size_t ws_size,
                              hipStream_t stream) {
    constexpr int N = 20000, E = 320000, HID = 256, NH = 8, NL = 4, FF = 1024, G = 32;

    // inputs in setup_inputs() dict order (params flattened in insertion order)
    const float* x      = (const float*)d_in[0];
    const int*   eidx   = (const int*)  d_in[1];
    const float* eattr  = (const float*)d_in[2];
    const int*   batch  = (const int*)  d_in[3];
    const float* in_w   = (const float*)d_in[4];
    const float* in_b   = (const float*)d_in[5];
    const float* Wq     = (const float*)d_in[6];
    const float* Wk     = (const float*)d_in[7];
    const float* Wv     = (const float*)d_in[8];
    const float* ee_w1  = (const float*)d_in[9];
    const float* ee_b1  = (const float*)d_in[10];
    const float* ee_w2  = (const float*)d_in[11];
    const float* ee_b2  = (const float*)d_in[12];
    const float* out_w  = (const float*)d_in[13];
    const float* out_b  = (const float*)d_in[14];
    const float* ln1_w  = (const float*)d_in[15];
    const float* ln1_b  = (const float*)d_in[16];
    const float* ln2_w  = (const float*)d_in[17];
    const float* ln2_b  = (const float*)d_in[18];
    const float* ffn_w1 = (const float*)d_in[19];
    const float* ffn_b1 = (const float*)d_in[20];
    const float* ffn_w2 = (const float*)d_in[21];
    const float* ffn_b2 = (const float*)d_in[22];
    const float* rb1_l1w = (const float*)d_in[23];
    const float* rb1_l1b = (const float*)d_in[24];
    const float* rb1_l2w = (const float*)d_in[25];
    const float* rb1_l2b = (const float*)d_in[26];
    const float* rb1_lnw = (const float*)d_in[27];
    const float* rb1_lnb = (const float*)d_in[28];
    const float* rb2_l1w = (const float*)d_in[29];
    const float* rb2_l1b = (const float*)d_in[30];
    const float* rb2_l2w = (const float*)d_in[31];
    const float* rb2_l2b = (const float*)d_in[32];
    const float* rb2_lnw = (const float*)d_in[33];
    const float* rb2_lnb = (const float*)d_in[34];
    const float* rb2_pw  = (const float*)d_in[35];
    const float* rb2_pb  = (const float*)d_in[36];
    const float* rb3_l1w = (const float*)d_in[37];
    const float* rb3_l1b = (const float*)d_in[38];
    const float* rb3_l2w = (const float*)d_in[39];
    const float* rb3_l2b = (const float*)d_in[40];
    const float* rb3_lnw = (const float*)d_in[41];
    const float* rb3_lnb = (const float*)d_in[42];
    const float* rb3_pw  = (const float*)d_in[43];
    const float* rb3_pb  = (const float*)d_in[44];
    const float* mean_w  = (const float*)d_in[45];
    const float* mean_b  = (const float*)d_in[46];
    const float* var_w   = (const float*)d_in[47];
    const float* var_b   = (const float*)d_in[48];

    const int* src = eidx;
    const int* dst = eidx + E;

    // workspace carve
    char* wsb = (char*)d_ws;
    size_t off = 0;
    auto carve = [&](size_t bytes) -> void* {
        void* p = wsb + off;
        off = (off + bytes + 255) & ~(size_t)255;
        return p;
    };
    float* h       = (float*)carve((size_t)N * HID * 4);
    h16_t* hh      = (h16_t*)carve((size_t)N * HID * 2);
    float* q       = (float*)carve((size_t)N * HID * 4);   // reused as agg
    float* k       = (float*)carve((size_t)N * HID * 4);   // reused as agg16
    float* v       = (float*)carve((size_t)N * HID * 4);   // reused as attn_out / ffn_out
    h16_t* mid16   = (h16_t*)carve((size_t)N * FF * 2);
    float* scores  = (float*)carve((size_t)E * NH * 4);    // overwritten with exp()
    float* gate    = (float*)carve((size_t)E * NH * 4);
    float* smax    = (float*)carve((size_t)N * NH * 4);
    float* den     = (float*)carve((size_t)N * NH * 4);
    h16_t* wq16    = (h16_t*)carve((size_t)NL * HID * HID * 2);
    h16_t* wk16    = (h16_t*)carve((size_t)NL * HID * HID * 2);
    h16_t* wv16    = (h16_t*)carve((size_t)NL * HID * HID * 2);
    h16_t* wo16    = (h16_t*)carve((size_t)NL * HID * HID * 2);
    h16_t* f1w16   = (h16_t*)carve((size_t)NL * FF * HID * 2);
    h16_t* f2w16   = (h16_t*)carve((size_t)NL * HID * FF * 2);
    float* emb_acc = (float*)carve((size_t)G * HID * 4);
    float* counts  = (float*)carve((size_t)G * 4);
    float* embbuf  = (float*)carve((size_t)G * HID * 4);
    float* sA      = (float*)carve((size_t)G * HID * 4);
    float* sB      = (float*)carve((size_t)G * HID * 4);
    float* sC      = (float*)carve((size_t)G * HID * 4);
    float* f1b     = (float*)carve((size_t)G * HID * 4);
    float* f2b     = (float*)carve((size_t)G * 128 * 4);
    float* f3b     = (float*)carve((size_t)G * 64 * 4);

    float* agg      = q;
    h16_t* agg16    = (h16_t*)k;
    float* attn_out = v;
    float* ffn_out  = v;

    float* out_mean = (float*)d_out;
    float* out_emb  = (float*)d_out + G;             // 32
    float* out_var  = (float*)d_out + G + G * HID;   // 32 + 8192

    const int T = 256;
    auto cdiv = [](int a, int b) { return (a + b - 1) / b; };
    auto gemm_grid = [&](int M, int OUT) { return cdiv((M / 16) * (OUT / 64), 8); };

    // weights -> f16 (once for all layers)
    convert_f16<<<cdiv(NL*HID*HID, T), T, 0, stream>>>(Wq,     wq16,  NL*HID*HID);
    convert_f16<<<cdiv(NL*HID*HID, T), T, 0, stream>>>(Wk,     wk16,  NL*HID*HID);
    convert_f16<<<cdiv(NL*HID*HID, T), T, 0, stream>>>(Wv,     wv16,  NL*HID*HID);
    convert_f16<<<cdiv(NL*HID*HID, T), T, 0, stream>>>(out_w,  wo16,  NL*HID*HID);
    convert_f16<<<cdiv(NL*FF*HID, T),  T, 0, stream>>>(ffn_w1, f1w16, NL*FF*HID);
    convert_f16<<<cdiv(NL*HID*FF, T),  T, 0, stream>>>(ffn_w2, f2w16, NL*HID*FF);

    input_proj<<<cdiv(N*HID, T), T, 0, stream>>>(x, in_w, in_b, h, hh, N*HID);

    for (int i = 0; i < NL; ++i) {
        // Q, K, V projections (no bias)
        gemm_wmma_f16<<<gemm_grid(N, HID), T, 0, stream>>>(hh, wq16 + (size_t)i*HID*HID, nullptr, q, nullptr, N, HID, HID, 0);
        gemm_wmma_f16<<<gemm_grid(N, HID), T, 0, stream>>>(hh, wk16 + (size_t)i*HID*HID, nullptr, k, nullptr, N, HID, HID, 0);
        gemm_wmma_f16<<<gemm_grid(N, HID), T, 0, stream>>>(hh, wv16 + (size_t)i*HID*HID, nullptr, v, nullptr, N, HID, HID, 0);

        // edge gating + attention scores + segment softmax
        edge_gate<<<cdiv(E, T), T, 0, stream>>>(eattr, ee_w1 + i*64, ee_b1 + i*64,
                                                ee_w2 + i*512, ee_b2 + i*8, gate, E);
        fill_f32<<<cdiv(N*NH, T), T, 0, stream>>>(smax, -INFINITY, N*NH);
        edge_scores<<<cdiv(E, 8), T, 0, stream>>>(q, k, src, dst, gate, scores, smax, E);
        fill_f32<<<cdiv(N*NH, T), T, 0, stream>>>(den, 0.0f, N*NH);
        edge_exp<<<cdiv(E*NH, T), T, 0, stream>>>(scores, dst, smax, den, E*NH);

        // weighted aggregation of v[src] into agg[dst]
        fill_f32<<<cdiv(N*HID, T), T, 0, stream>>>(agg, 0.0f, N*HID);
        edge_aggregate<<<cdiv(E, 8), T, 0, stream>>>(v, scores, den, src, dst, agg, E);

        // out projection + residual LN1
        convert_f16<<<cdiv(N*HID, T), T, 0, stream>>>(agg, agg16, N*HID);
        gemm_wmma_f16<<<gemm_grid(N, HID), T, 0, stream>>>(agg16, wo16 + (size_t)i*HID*HID,
                                                           out_b + i*HID, attn_out, nullptr, N, HID, HID, 0);
        ln_rows<<<cdiv(N, 8), T, 0, stream>>>(h, attn_out, ln1_w + i*HID, ln1_b + i*HID, h, hh, N, HID);

        // FFN (gelu fused in gemm1 epilogue) + residual LN2
        gemm_wmma_f16<<<gemm_grid(N, FF), T, 0, stream>>>(hh, f1w16 + (size_t)i*FF*HID,
                                                          ffn_b1 + i*FF, nullptr, mid16, N, HID, FF, 1);
        gemm_wmma_f16<<<gemm_grid(N, HID), T, 0, stream>>>(mid16, f2w16 + (size_t)i*HID*FF,
                                                           ffn_b2 + i*HID, ffn_out, nullptr, N, FF, HID, 0);
        ln_rows<<<cdiv(N, 8), T, 0, stream>>>(h, ffn_out, ln2_w + i*HID, ln2_b + i*HID, h, hh, N, HID);
    }

    // global mean pool per graph
    fill_f32<<<cdiv(G*HID, T), T, 0, stream>>>(emb_acc, 0.0f, G*HID);
    fill_f32<<<1, G, 0, stream>>>(counts, 0.0f, G);
    pool_count<<<cdiv(N, T), T, 0, stream>>>(batch, counts, N);
    pool_accum<<<cdiv(N*HID, T), T, 0, stream>>>(h, batch, emb_acc, N*HID);
    pool_final<<<cdiv(G*HID, T), T, 0, stream>>>(emb_acc, counts, out_emb, embbuf, G*HID);

    // residual block 1 (256 -> 256, identity residual)
    small_linear<<<cdiv(G*256, T), T, 0, stream>>>(embbuf, rb1_l1w, rb1_l1b, sA, G, 256, 256, 1);
    small_linear<<<cdiv(G*256, T), T, 0, stream>>>(sA, rb1_l2w, rb1_l2b, sB, G, 256, 256, 0);
    ln_rows<<<cdiv(G, 8), T, 0, stream>>>(sB, embbuf, rb1_lnw, rb1_lnb, f1b, nullptr, G, 256);
    // residual block 2 (256 -> 128, projected residual)
    small_linear<<<cdiv(G*128, T), T, 0, stream>>>(f1b, rb2_pw, rb2_pb, sC, G, 256, 128, 0);
    small_linear<<<cdiv(G*128, T), T, 0, stream>>>(f1b, rb2_l1w, rb2_l1b, sA, G, 256, 128, 1);
    small_linear<<<cdiv(G*128, T), T, 0, stream>>>(sA, rb2_l2w, rb2_l2b, sB, G, 128, 128, 0);
    ln_rows<<<cdiv(G, 8), T, 0, stream>>>(sB, sC, rb2_lnw, rb2_lnb, f2b, nullptr, G, 128);
    // residual block 3 (128 -> 64, projected residual)
    small_linear<<<cdiv(G*64, T), T, 0, stream>>>(f2b, rb3_pw, rb3_pb, sC, G, 128, 64, 0);
    small_linear<<<cdiv(G*64, T), T, 0, stream>>>(f2b, rb3_l1w, rb3_l1b, sA, G, 128, 64, 1);
    small_linear<<<cdiv(G*64, T), T, 0, stream>>>(sA, rb3_l2w, rb3_l2b, sB, G, 64, 64, 0);
    ln_rows<<<cdiv(G, 8), T, 0, stream>>>(sB, sC, rb3_lnw, rb3_lnb, f3b, nullptr, G, 64);

    // heads: mean + var
    heads_kernel<<<1, 32, 0, stream>>>(f3b, mean_w, mean_b, var_w, var_b, out_mean, out_var);

    (void)in_sizes; (void)n_in; (void)out_size; (void)ws_size;
}
